// InfoNCELoss_3573412790311
// MI455X (gfx1250) — compile-verified, matrix operand order checked by reading, not run on hardware
//
#include <hip/hip_runtime.h>

// ---------- types ----------
typedef __bf16 bf16_t;
typedef __attribute__((ext_vector_type(4)))  __bf16 v4bf;
typedef __attribute__((ext_vector_type(8)))  __bf16 v8bf;
typedef __attribute__((ext_vector_type(16))) __bf16 v16bf;
typedef __attribute__((ext_vector_type(8)))  float  v8f;

// ---------- problem constants ----------
#define NROWS 8192
#define DDIM  1024
#define INV_T 14.2857142857142857f   // 1/0.07
#define NRM_EPS 1e-12f

// ---------- GEMM tile config ----------
#define BM 128
#define BN 128
#define BK 32
#define KSTEPS (DDIM / BK)           // 32 (power of two)
#define LDA 40   // padded LDS row stride in bf16 (80B rows -> conflict-free)

// =====================================================================
// CDNA5 async global->LDS copy (ASYNCcnt-tracked DMA, bypasses VGPRs).
// INST_OFFSET is added to BOTH the LDS and global addresses, so one
// lds-addr VGPR + one 32-bit global offset covers 2 chunks (offset 0/16).
// =====================================================================
__device__ __forceinline__ void async_b128x2(unsigned lds_addr, unsigned gvoff,
                                             const void* sbase) {
    asm volatile(
        "global_load_async_to_lds_b128 %0, %1, %2 offset:0\n\t"
        "global_load_async_to_lds_b128 %0, %1, %2 offset:16"
        :
        : "v"(lds_addr), "v"(gvoff), "s"(sbase)
        : "memory");
}

__device__ __forceinline__ void wait_asynccnt_le4() {
    asm volatile("s_wait_asynccnt 0x4" ::: "memory");
}

// =====================================================================
// Kernel 1: zero fp32 accumulators (row_sum ++ col_sum, contiguous)
// =====================================================================
__global__ void infonce_zero_kernel(float* __restrict__ p, int n) {
    int i = blockIdx.x * blockDim.x + threadIdx.x;
    if (i < n) p[i] = 0.0f;
}

// =====================================================================
// Kernel 2: fused L2-normalize (fp32 in) -> bf16 out. One wave per row.
// =====================================================================
__global__ __launch_bounds__(256)
void infonce_normalize_kernel(const float* __restrict__ q,
                              const float* __restrict__ k,
                              bf16_t* __restrict__ qb,
                              bf16_t* __restrict__ kb) {
    const int wave = threadIdx.x >> 5;
    const int lane = threadIdx.x & 31;
    const int r = blockIdx.x * 8 + wave;          // 0..2N-1

    const float* src;
    bf16_t* dst;
    if (r < NROWS) { src = q + (size_t)r * DDIM;           dst = qb + (size_t)r * DDIM; }
    else           { src = k + (size_t)(r - NROWS) * DDIM; dst = kb + (size_t)(r - NROWS) * DDIM; }

    const float4* s4 = (const float4*)src;
    float4 v[8];
    float ss = 0.0f;
#pragma unroll
    for (int j = 0; j < 8; ++j) {
        v[j] = s4[lane + 32 * j];
        ss += v[j].x * v[j].x + v[j].y * v[j].y + v[j].z * v[j].z + v[j].w * v[j].w;
    }
#pragma unroll
    for (int m = 16; m >= 1; m >>= 1) ss += __shfl_xor(ss, m, 32);

    const float inv = 1.0f / fmaxf(sqrtf(ss), NRM_EPS);

#pragma unroll
    for (int j = 0; j < 8; ++j) {
        v4bf o;
        o[0] = (bf16_t)(v[j].x * inv);
        o[1] = (bf16_t)(v[j].y * inv);
        o[2] = (bf16_t)(v[j].z * inv);
        o[3] = (bf16_t)(v[j].w * inv);
        *(v4bf*)(dst + (size_t)(lane + 32 * j) * 4) = o;
    }
}

// =====================================================================
// Kernel 3: diagonal logits. diag[n] = (qn[n] . kn[n]) * (1/T).
// =====================================================================
__global__ __launch_bounds__(256)
void infonce_diag_kernel(const bf16_t* __restrict__ qb,
                         const bf16_t* __restrict__ kb,
                         float* __restrict__ diag) {
    const int wave = threadIdx.x >> 5;
    const int lane = threadIdx.x & 31;
    const int n = blockIdx.x * 8 + wave;

    const bf16_t* qr = qb + (size_t)n * DDIM + lane * 32;
    const bf16_t* kr = kb + (size_t)n * DDIM + lane * 32;

    float s = 0.0f;
#pragma unroll
    for (int j = 0; j < 4; ++j) {
        v8bf a = *(const v8bf*)(qr + j * 8);
        v8bf b = *(const v8bf*)(kr + j * 8);
#pragma unroll
        for (int i = 0; i < 8; ++i) s += (float)a[i] * (float)b[i];
    }
#pragma unroll
    for (int m = 16; m >= 1; m >>= 1) s += __shfl_xor(s, m, 32);

    if (lane == 0) diag[n] = s * INV_T;
}

// =====================================================================
// Kernel 4: double-buffered async-copy bf16 WMMA GEMM with fused
// shifted-exp LSE accumulation.
// Block: 256 threads = 8 waves (4 m-waves x 2 n-waves), tile 128x128.
// Each wave: 32x64 output = 2x4 v_wmma_f32_16x16x32_bf16 accumulators.
// Logits bounded by 1/T -> fixed shift C = 1/T makes LSE additive:
//   row_sum[m] += sum_n exp(l - C) ; col_sum[n] += sum_m exp(l - C)
// =====================================================================
__device__ __forceinline__ v16bf frag16(const bf16_t* p0, const bf16_t* p1) {
    v8bf lo = *(const v8bf*)p0;
    v8bf hi = *(const v8bf*)p1;
    return __builtin_shufflevector(lo, hi, 0, 1, 2, 3, 4, 5, 6, 7,
                                           8, 9, 10, 11, 12, 13, 14, 15);
}

// One pipeline stage: issue DMA for the NEXT slab into the other buffer,
// retire the CURRENT slab's 4 DMAs, sync, then 8 WMMAs from LDS.
__device__ __forceinline__ void gemm_step(const bf16_t* __restrict__ Ab,
                                          const bf16_t* __restrict__ Bb,
                                          unsigned nextAL, unsigned nextBL,
                                          unsigned koff,
                                          const bf16_t* __restrict__ qb,
                                          const bf16_t* __restrict__ kb,
                                          unsigned aG0, unsigned bG0,
                                          int mw, int nw, int lr, int hi,
                                          v8f acc[2][4]) {
    async_b128x2(nextAL, aG0 + koff, qb);
    async_b128x2(nextBL, bG0 + koff, kb);
    wait_asynccnt_le4();     // current slab complete (in-order retire)
    __syncthreads();

    // Fragments: lane lr = row/col within 16-tile; K striped as
    // {hi*8..+7} in v0-3 and {16+hi*8..} in v4-7 (16-bit layout).
    // D = A x B^T with both row-major => A and B gather identically.
    v16bf af[2], bfg[4];
#pragma unroll
    for (int mt = 0; mt < 2; ++mt) {
        const bf16_t* p = Ab + (mw * 32 + mt * 16 + lr) * LDA + hi * 8;
        af[mt] = frag16(p, p + 16);
    }
#pragma unroll
    for (int nt = 0; nt < 4; ++nt) {
        const bf16_t* p = Bb + (nw * 64 + nt * 16 + lr) * LDA + hi * 8;
        bfg[nt] = frag16(p, p + 16);
    }
#pragma unroll
    for (int mt = 0; mt < 2; ++mt)
#pragma unroll
        for (int nt = 0; nt < 4; ++nt)
            acc[mt][nt] = __builtin_amdgcn_wmma_f32_16x16x32_bf16(
                false, af[mt], false, bfg[nt], (short)0, acc[mt][nt],
                false, false);

    __syncthreads();   // all waves done reading this buffer before overwrite
}

__global__ __launch_bounds__(256)
void infonce_gemm_lse_kernel(const bf16_t* __restrict__ qb,
                             const bf16_t* __restrict__ kb,
                             float* __restrict__ row_sum,
                             float* __restrict__ col_sum) {
    __shared__ __align__(16) bf16_t As[2][BM * LDA];
    __shared__ __align__(16) bf16_t Bs[2][BN * LDA];
    __shared__ float rAcc[BM];
    __shared__ float cAcc[BN];

    const int tid  = threadIdx.x;
    const int lane = tid & 31;
    const int wv   = tid >> 5;
    const int mw   = wv >> 1;          // 0..3  (wave tile row base = mw*32)
    const int nw   = wv & 1;           // 0..1  (wave tile col base = nw*64)
    const int lr   = lane & 15;
    const int hi   = lane >> 4;        // 0 or 1

    const int bm = blockIdx.y * BM;
    const int bn = blockIdx.x * BN;

    // staging: 2 threads per row, 16 contiguous bf16 (32B) each, A and B
    const int sr = tid >> 1, sc = (tid & 1) * 16;

    // 32-bit global byte offsets at k=0 (fits: 8192*1024*2 = 16 MB)
    const unsigned aG0 = (unsigned)(((bm + sr) * DDIM + sc) * 2);
    const unsigned bG0 = (unsigned)(((bn + sr) * DDIM + sc) * 2);

    // LDS byte addresses of this thread's staging slot, per buffer
    const unsigned aL0 = (unsigned)(uintptr_t)&As[0][sr * LDA + sc];
    const unsigned aL1 = (unsigned)(uintptr_t)&As[1][sr * LDA + sc];
    const unsigned bL0 = (unsigned)(uintptr_t)&Bs[0][sr * LDA + sc];
    const unsigned bL1 = (unsigned)(uintptr_t)&Bs[1][sr * LDA + sc];

    v8f acc[2][4] = {};

    // prologue: DMA slab 0 into buffer 0 (4 async b128 per thread)
    async_b128x2(aL0, aG0, qb);
    async_b128x2(bL0, bG0, kb);

    // K loop unrolled by 2: straight-line double buffering, no cur-branch.
#pragma unroll 1
    for (int s = 0; s < KSTEPS; s += 2) {
        const unsigned k1 = (unsigned)((s + 1) * BK * 2);
        const unsigned k2 = (unsigned)(((s + 2) & (KSTEPS - 1)) * BK * 2);
        gemm_step(As[0], Bs[0], aL1, bL1, k1, qb, kb, aG0, bG0,
                  mw, nw, lr, hi, acc);
        gemm_step(As[1], Bs[1], aL0, bL0, k2, qb, kb, aG0, bG0,
                  mw, nw, lr, hi, acc);
    }

    // ---- fused epilogue: shifted exp + row/col accumulation ----
    if (tid < BM) rAcc[tid] = 0.0f;
    if (tid < BN) cAcc[tid] = 0.0f;
    __syncthreads();

    // C/D layout: VGPR r, lanes 0-15 -> (M=r, N=lr); lanes 16-31 -> (M=8+r, N=lr)
#pragma unroll
    for (int mt = 0; mt < 2; ++mt) {
#pragma unroll
        for (int nt = 0; nt < 4; ++nt) {
            float cp = 0.0f;
#pragma unroll
            for (int r = 0; r < 8; ++r) {
                float e = __expf(fmaf(acc[mt][nt][r], INV_T, -INV_T));
                cp += e;
                // xor masks < 16 stay inside each 16-lane half at width 32
                float rs = e;
                rs += __shfl_xor(rs, 1, 32);
                rs += __shfl_xor(rs, 2, 32);
                rs += __shfl_xor(rs, 4, 32);
                rs += __shfl_xor(rs, 8, 32);
                if (lr == 0)
                    atomicAdd(&rAcc[mw * 32 + mt * 16 + hi * 8 + r], rs);
            }
            atomicAdd(&cAcc[nw * 64 + nt * 16 + lr], cp);
        }
    }
    __syncthreads();

    if (tid < BM) atomicAdd(&row_sum[bm + tid], rAcc[tid]);
    if (tid < BN) atomicAdd(&col_sum[bn + tid], cAcc[tid]);
}

// =====================================================================
// Kernel 5: finalize.
// loss = 1/T + (1/N) * sum_n [ 0.5*(log rs[n] + log cs[n]) - diag[n] ]
// =====================================================================
__global__ __launch_bounds__(256)
void infonce_finalize_kernel(const float* __restrict__ row_sum,
                             const float* __restrict__ col_sum,
                             const float* __restrict__ diag,
                             float* __restrict__ out) {
    __shared__ float red[256];
    float s = 0.0f;
    for (int i = threadIdx.x; i < NROWS; i += 256)
        s += 0.5f * (logf(row_sum[i]) + logf(col_sum[i])) - diag[i];
    red[threadIdx.x] = s;
    __syncthreads();
    for (int off = 128; off > 0; off >>= 1) {
        if (threadIdx.x < off) red[threadIdx.x] += red[threadIdx.x + off];
        __syncthreads();
    }
    if (threadIdx.x == 0) out[0] = INV_T + red[0] * (1.0f / (float)NROWS);
}

// =====================================================================
// Host-side launcher
// =====================================================================
extern "C" void kernel_launch(void* const* d_in, const int* in_sizes, int n_in,
                              void* d_out, int out_size, void* d_ws, size_t ws_size,
                              hipStream_t stream) {
    (void)in_sizes; (void)n_in; (void)out_size; (void)ws_size;

    const float* q = (const float*)d_in[0];
    const float* k = (const float*)d_in[1];
    float* out = (float*)d_out;

    // workspace layout
    char* ws = (char*)d_ws;
    bf16_t* qb = (bf16_t*)ws;                                   // 16 MB
    bf16_t* kb = qb + (size_t)NROWS * DDIM;                     // 16 MB
    float* row_sum = (float*)(ws + (size_t)2 * NROWS * DDIM * sizeof(bf16_t));
    float* col_sum = row_sum + NROWS;
    float* diag    = col_sum + NROWS;

    // 1) zero row/col accumulators (contiguous 2N floats)
    infonce_zero_kernel<<<(2 * NROWS + 255) / 256, 256, 0, stream>>>(row_sum, 2 * NROWS);

    // 2) normalize both matrices -> bf16 (one wave per row, 2N rows)
    infonce_normalize_kernel<<<(2 * NROWS) / 8, 256, 0, stream>>>(q, k, qb, kb);

    // 3) diagonal logits
    infonce_diag_kernel<<<NROWS / 8, 256, 0, stream>>>(qb, kb, diag);

    // 4) fused async-copy WMMA GEMM + shifted-exp row/col accumulation
    dim3 grid(NROWS / BN, NROWS / BM);
    infonce_gemm_lse_kernel<<<grid, 256, 0, stream>>>(qb, kb, row_sum, col_sum);

    // 5) finalize to scalar
    infonce_finalize_kernel<<<1, 256, 0, stream>>>(row_sum, col_sum, diag, out);
}
